// Network_Search_38560216383904
// MI455X (gfx1250) — matrix operand order repaired and checked by reading, not compiled, not run on hardware
//
#include <hip/hip_runtime.h>
#include <math.h>

typedef __attribute__((ext_vector_type(16))) _Float16 v16h;
typedef __attribute__((ext_vector_type(8)))  _Float16 v8h;
typedef __attribute__((ext_vector_type(8)))  float    v8f;
typedef __attribute__((ext_vector_type(4)))  unsigned int v4u;
typedef __attribute__((ext_vector_type(8)))  int      v8i;
typedef __attribute__((ext_vector_type(4)))  int      v4i;

#define NCOL 22
#define NEMB 12
#define DD   64
#define BB   2048
#define KTOT (NCOL*DD)          // 1408 = 44 * 32
#define NPAD 16
#define REGC 0.001f
#define NPAIR (NCOL*NCOL)       // 484
#define PSTRIDE (NPAIR*2)       // 968 floats per d-slice per array
#define NCHUNK 8                // d split 8 ways (8 d per chunk)

#if __has_builtin(__builtin_amdgcn_tensor_load_to_lds)
#define HAVE_TDM 1
#else
#define HAVE_TDM 0
#endif

// ---------------------------------------------------------------------------
// Kernel 1: embedding gather + tiny MLP -> trans (f16 GEMM copy + f32 copy),
//           plus per-block partial sums of E^2 per column (deterministic).
// grid = (512, 22), block = 256  (4 batch rows x 64 d per block)
// ---------------------------------------------------------------------------
__global__ void __launch_bounds__(256)
k_embed_mlp(const int* __restrict__ feats, const float* __restrict__ tables,
            const float* __restrict__ w1, const float* __restrict__ b1,
            const float* __restrict__ w2, const float* __restrict__ b2,
            _Float16* __restrict__ transH, float* __restrict__ transF,
            float* __restrict__ partial)
{
    const int i  = blockIdx.y;
    const int bt = blockIdx.x;
    const int t  = threadIdx.x;
    const int bl = t >> 6;          // 0..3
    const int d  = t & 63;
    const int b  = bt * 4 + bl;

    const int   f = feats[i * BB + b];
    const float e = tables[(i * NEMB + f) * DD + d];

    float acc = b2[0];
    #pragma unroll
    for (int k = 0; k < 8; ++k)
        acc += w2[k] * tanhf(e * w1[k] + b1[k]);

    transH[b * KTOT + i * DD + d]   = (_Float16)acc;   // A matrix: [b][i*64+d]
    transF[(b * DD + d) * NCOL + i] = acc;             // pairwise: [b][d][i]

    // block-level reduction of e*e (fixed tree order -> deterministic)
    __shared__ float red[256];
    red[t] = e * e;
    __syncthreads();
    for (int s = 128; s > 0; s >>= 1) {
        if (t < s) red[t] += red[t + s];
        __syncthreads();
    }
    if (t == 0) partial[i * 512 + bt] = red[0];
}

// ---------------------------------------------------------------------------
// Kernel 2: fold all rank-1 weight terms into the f16 B-matrix, stored
// transposed: Bt[n][k], n in [0,16) (only n<2 nonzero), k = i*64+d.
// ---------------------------------------------------------------------------
__global__ void __launch_bounds__(256)
k_prep_lin(const float* __restrict__ W_ops, const float* __restrict__ W_cat,
           const float* __restrict__ aw, _Float16* __restrict__ Bt)
{
    int tid = blockIdx.x * 256 + threadIdx.x;
    if (tid >= NPAD * KTOT) return;
    int n = tid / KTOT, k = tid % KTOT;
    if (n >= 2) { Bt[tid] = (_Float16)0.0f; return; }
    int i = k / DD, d = k % DD, o = n;

    float rp = 0.f, cp = 0.f, rmx = 0.f, cmx = 0.f, rmn = 0.f, cmn = 0.f;
    float rc1 = 0.f, cc2 = 0.f;
    for (int j = 0; j < NCOL; ++j) {
        rp  += W_ops[(((0 * NCOL + i) * NCOL + j) * 2 + o) * DD + d];
        cp  += W_ops[(((0 * NCOL + j) * NCOL + i) * 2 + o) * DD + d];
        rmx += W_ops[(((2 * NCOL + i) * NCOL + j) * 2 + o) * DD + d];
        cmx += W_ops[(((2 * NCOL + j) * NCOL + i) * 2 + o) * DD + d];
        rmn += W_ops[(((3 * NCOL + i) * NCOL + j) * 2 + o) * DD + d];
        cmn += W_ops[(((3 * NCOL + j) * NCOL + i) * 2 + o) * DD + d];
        rc1 += W_cat[((i * NCOL + j) * 2 + o) * (2 * DD) + d];         // Wc1 row-sum
        cc2 += W_cat[((j * NCOL + i) * 2 + o) * (2 * DD) + DD + d];    // Wc2 col-sum
    }
    float v = aw[0] * (rp + cp) + aw[4] * (rc1 + cc2)
            + 0.5f * aw[2] * (rmx + cmx) + 0.5f * aw[3] * (rmn + cmn);
    Bt[tid] = (_Float16)v;
}

// ---------------------------------------------------------------------------
// Kernel 2b: pairwise weights, pre-scaled and reordered to [d][(i*22+j)*2+o]:
//   WmS = aw1 * Wm ; Wd = 0.5*(aw2*Wmax - aw3*Wmin)
// ---------------------------------------------------------------------------
__global__ void __launch_bounds__(256)
k_prep_pair(const float* __restrict__ W_ops, const float* __restrict__ aw,
            float* __restrict__ WmS, float* __restrict__ Wd)
{
    int tid = blockIdx.x * 256 + threadIdx.x;
    if (tid >= DD * PSTRIDE) return;
    int d = tid / PSTRIDE;
    int p = tid % PSTRIDE;
    int o = p & 1, pair = p >> 1;
    int i = pair / NCOL, j = pair % NCOL;
    float wm  = W_ops[(((1 * NCOL + i) * NCOL + j) * 2 + o) * DD + d];
    float wmx = W_ops[(((2 * NCOL + i) * NCOL + j) * 2 + o) * DD + d];
    float wmn = W_ops[(((3 * NCOL + i) * NCOL + j) * 2 + o) * DD + d];
    WmS[tid] = aw[1] * wm;
    Wd[tid]  = 0.5f * (aw[2] * wmx - aw[3] * wmn);
}

// ---------------------------------------------------------------------------
// TDM helper: issue a 2D tile load (width elements x rows, 4-byte elements,
// row pitch in elements) from global into contiguous LDS at lds_addr.
// D# packing per CDNA5 ISA 8.3/8.4. Wave-level op; EXEC ignored.
// ---------------------------------------------------------------------------
#if HAVE_TDM
__device__ static inline void tdm_load_2d(unsigned lds_addr, const void* gaddr,
                                          unsigned width_elts, unsigned rows,
                                          unsigned row_pitch_elts)
{
    unsigned long long ga = (unsigned long long)(uintptr_t)gaddr;
    v4u g0;
    g0[0] = 1u;                                        // count=1, user mode
    g0[1] = lds_addr;                                  // lds_addr (bytes)
    g0[2] = (unsigned)(ga & 0xFFFFFFFFu);              // global_addr[31:0]
    g0[3] = (unsigned)((ga >> 32) & 0x01FFFFFFu)       // global_addr[56:32]
          | (2u << 30);                                // type = 2 ("image")
    v8i g1;
    g1[0] = (int)(2u << 16);                           // data_size=2 (4 bytes)
    g1[1] = (int)((width_elts & 0xFFFFu) << 16);       // tensor_dim0[15:0]
    g1[2] = (int)((width_elts >> 16) | ((rows & 0xFFFFu) << 16)); // dim0 hi | dim1 lo
    g1[3] = (int)((width_elts & 0xFFFFu) << 16);       // tile_dim0
    g1[4] = (int)(rows & 0xFFFFu);                     // tile_dim1
    g1[5] = (int)row_pitch_elts;                       // tensor_dim0_stride[31:0]
    g1[6] = 0;
    g1[7] = 0;
    v4i z4 = {0, 0, 0, 0};
#if defined(__clang_major__) && (__clang_major__ >= 23)
    v8i z8 = {0, 0, 0, 0, 0, 0, 0, 0};
    __builtin_amdgcn_tensor_load_to_lds(g0, g1, z4, z4, z8, 0);
#else
    __builtin_amdgcn_tensor_load_to_lds(g0, g1, z4, z4, 0);
#endif
}
#endif

// ---------------------------------------------------------------------------
// Kernel 3: genuinely-pairwise part, d split 8 ways per batch row.
// thread = (chunk = t>>5, b_local = t&31); block covers 32 b x 8 chunks.
// All 8 per-chunk weight slices (60.5 KB) staged in LDS via TDM (or
// cooperative loads), then deterministic tree-reduction over chunks.
// grid = 64, block = 256  -> 512 waves of real VALU work.
// ---------------------------------------------------------------------------
__global__ void __launch_bounds__(256)
k_pairwise(const float* __restrict__ transF, const float* __restrict__ WmS,
           const float* __restrict__ Wd, float* __restrict__ out_pair)
{
    __shared__ float sWm[NCHUNK][PSTRIDE];   // 8 x 968 floats
    __shared__ float sWd[NCHUNK][PSTRIDE];
    const int t     = threadIdx.x;
    const int bl    = t & 31;                // b within block
    const int chunk = t >> 5;                // 0..7 (== wave id)
    const int b     = blockIdx.x * 32 + bl;

    float a0 = 0.f, a1 = 0.f;
    for (int dl = 0; dl < 8; ++dl) {
        __syncthreads();                     // previous slice fully consumed
#if HAVE_TDM
        if (t == 0) {
            // chunk c needs global row d = c*8 + dl -> start row dl, pitch 8 rows
            tdm_load_2d((unsigned)(uintptr_t)&sWm[0][0],
                        WmS + (size_t)dl * PSTRIDE, PSTRIDE, NCHUNK,
                        NCHUNK * PSTRIDE);
            tdm_load_2d((unsigned)(uintptr_t)&sWd[0][0],
                        Wd + (size_t)dl * PSTRIDE, PSTRIDE, NCHUNK,
                        NCHUNK * PSTRIDE);
            __builtin_amdgcn_s_wait_tensorcnt(0);
        }
#else
        for (int idx = t; idx < NCHUNK * PSTRIDE; idx += 256) {
            int c = idx / PSTRIDE, w = idx % PSTRIDE;
            sWm[c][w] = WmS[(size_t)(c * 8 + dl) * PSTRIDE + w];
            sWd[c][w] = Wd [(size_t)(c * 8 + dl) * PSTRIDE + w];
        }
#endif
        __syncthreads();                     // slice published to all waves

        const int d = chunk * 8 + dl;
        const float* tp = transF + (size_t)(b * DD + d) * NCOL;
        float tv[NCOL];
        #pragma unroll
        for (int i = 0; i < NCOL; ++i) tv[i] = tp[i];

        #pragma unroll
        for (int i = 0; i < NCOL; ++i) {
            const float ti = tv[i];
            #pragma unroll
            for (int j = 0; j < NCOL; ++j) {
                const float tj = tv[j];
                const float pr = ti * tj;
                const float ad = fabsf(ti - tj);
                const int base = (i * NCOL + j) * 2;
                a0 += pr * sWm[chunk][base]     + ad * sWd[chunk][base];
                a1 += pr * sWm[chunk][base + 1] + ad * sWd[chunk][base + 1];
            }
        }
    }

    // deterministic reduction across chunks (stride-32 lanes), reusing LDS
    float* r0 = &sWm[0][0];
    float* r1 = &sWm[0][0] + 256;
    __syncthreads();
    r0[t] = a0; r1[t] = a1;
    __syncthreads();
    for (int s = 128; s >= 32; s >>= 1) {
        if (t < s) { r0[t] += r0[t + s]; r1[t] += r1[t + s]; }
        __syncthreads();
    }
    if (t < 32) {
        const int bo = blockIdx.x * 32 + t;
        out_pair[bo * 2]     = r0[t];
        out_pair[bo * 2 + 1] = r1[t];
    }
}

// ---------------------------------------------------------------------------
// Kernel 4: WMMA GEMM. C[2048,16] = A[2048,1408]_f16 x Bt^T, K = 44*32.
// Register layouts per CDNA5 ISA 7.12.2 (see round-1 comments).
// grid = 16, block = 256 (8 waves, one 16-row tile each)
// ---------------------------------------------------------------------------
__global__ void __launch_bounds__(256)
k_gemm_wmma(const _Float16* __restrict__ A, const _Float16* __restrict__ Bt,
            const float* __restrict__ out_pair, float* __restrict__ out)
{
    const int lane  = threadIdx.x & 31;
    const int wave  = threadIdx.x >> 5;
    const int btile = blockIdx.x * 8 + wave;     // 0..127
    const int m     = lane & 15;
    const int hi    = lane >> 4;

    const _Float16* arow = A  + (size_t)(btile * 16 + m) * KTOT + hi * 8;
    const _Float16* brow = Bt + (size_t)m * KTOT + hi * 8;

    v8f acc = {};
    for (int kk = 0; kk < KTOT; kk += 32) {
        __builtin_prefetch(arow + kk + 64, 0, 1);
        v8h A0 = *(const v8h*)(arow + kk);        // K = kk + 8*hi + [0..7]
        v8h A1 = *(const v8h*)(arow + kk + 16);   // K = kk + 16 + 8*hi + [0..7]
        v8h B0 = *(const v8h*)(brow + kk);        // K = kk + 8*hi + [0..7]
        v8h B1 = *(const v8h*)(brow + kk + 16);   // K = kk + 16 + 8*hi + [0..7]
        v16h av, bv;
        #pragma unroll
        for (int e = 0; e < 8; ++e) {
            av[e] = A0[e]; av[e + 8] = A1[e];
            bv[e] = B0[e]; bv[e + 8] = B1[e];
        }
        acc = __builtin_amdgcn_wmma_f32_16x16x32_f16(
                  /*neg_a=*/false, av, /*neg_b=*/false, bv,
                  /*c_mod=*/(short)0, acc, /*reuse_a=*/false, /*reuse_b=*/false);
    }

    const int n = m;                 // C column = lane % 16; only o<2 live
    if (n < 2) {
        #pragma unroll
        for (int r = 0; r < 8; ++r) {
            const int row = btile * 16 + hi * 8 + r;
            out[row * 2 + n] = acc[r] + out_pair[row * 2 + n];
        }
    }
}

// ---------------------------------------------------------------------------
// Kernel 5: regularizer, fixed serial order (deterministic).
// ---------------------------------------------------------------------------
__global__ void k_regs(const float* __restrict__ partial, float* __restrict__ out)
{
    if (threadIdx.x == 0 && blockIdx.x == 0) {
        float tot = 0.f;
        for (int i = 0; i < NCOL; ++i) {
            float s = 0.f;
            for (int blk = 0; blk < 512; ++blk) s += partial[i * 512 + blk];
            tot += sqrtf(s);
        }
        out[BB * 2] = REGC * (2.0f * NCOL) * tot;
    }
}

// ---------------------------------------------------------------------------
extern "C" void kernel_launch(void* const* d_in, const int* in_sizes, int n_in,
                              void* d_out, int out_size, void* d_ws, size_t ws_size,
                              hipStream_t stream)
{
    (void)in_sizes; (void)n_in; (void)out_size; (void)ws_size;

    const int*   feats  = (const int*)  d_in[0];
    const float* tables = (const float*)d_in[1];
    const float* w1     = (const float*)d_in[2];
    const float* b1     = (const float*)d_in[3];
    const float* w2     = (const float*)d_in[4];
    const float* b2     = (const float*)d_in[5];
    const float* W_ops  = (const float*)d_in[6];
    const float* W_cat  = (const float*)d_in[7];
    const float* aw     = (const float*)d_in[8];
    float* out = (float*)d_out;

    // workspace layout (256B-aligned slabs)
    char* ws = (char*)d_ws;
    size_t off = 0;
    auto slab = [&](size_t bytes) {
        char* p = ws + off;
        off = (off + bytes + 255) & ~(size_t)255;
        return p;
    };
    _Float16* transH   = (_Float16*)slab((size_t)BB * KTOT * 2);          // 5.8 MB
    float*    transF   = (float*)   slab((size_t)BB * DD * NCOL * 4);     // 11.5 MB
    _Float16* Bt       = (_Float16*)slab((size_t)NPAD * KTOT * 2);        // 45 KB
    float*    WmS      = (float*)   slab((size_t)DD * PSTRIDE * 4);       // 248 KB
    float*    Wd       = (float*)   slab((size_t)DD * PSTRIDE * 4);       // 248 KB
    float*    out_pair = (float*)   slab((size_t)BB * 2 * 4);             // 16 KB
    float*    partial  = (float*)   slab((size_t)NCOL * 512 * 4);         // 45 KB

    // 1) gather + MLP + norm partials
    k_embed_mlp<<<dim3(512, NCOL), 256, 0, stream>>>(
        feats, tables, w1, b1, w2, b2, transH, transF, partial);

    // 2) fold rank-1 terms into f16 B-matrix (transposed, N padded to 16)
    k_prep_lin<<<(NPAD * KTOT + 255) / 256, 256, 0, stream>>>(W_ops, W_cat, aw, Bt);

    // 2b) scaled pairwise weights reordered [d][pair][o]
    k_prep_pair<<<(DD * PSTRIDE + 255) / 256, 256, 0, stream>>>(W_ops, aw, WmS, Wd);

    // 3) pairwise (quadratic + |diff|) part, TDM-staged weights, 512 waves
    k_pairwise<<<BB / 32, 256, 0, stream>>>(transF, WmS, Wd, out_pair);

    // 4) WMMA GEMM (M=2048,N=16,K=1408) + add pairwise -> final inferences
    k_gemm_wmma<<<16, 256, 0, stream>>>(transH, Bt, out_pair, out);

    // 5) regularizer scalar
    k_regs<<<1, 64, 0, stream>>>(partial, out);
}
